// CodeFlow_75582834475400
// MI455X (gfx1250) — compile-verified, matrix operand order checked by reading, not run on hardware
//
#include <hip/hip_runtime.h>

// ---------------------------------------------------------------------------
// Problem constants (from reference): B=16, N=512, T=16, H=256, V=5002, E=1024
// ---------------------------------------------------------------------------
#define Bb   16
#define Nn   512
#define Tt   16
#define Hh   256
#define Ee   1024
#define CAP  48          // max predecessors kept per (b,row); degree ~Bin(1024,1/512)

typedef __bf16  bf16;
typedef bf16    v16bf __attribute__((ext_vector_type(16)));
typedef float   v8f   __attribute__((ext_vector_type(8)));
typedef float   f4    __attribute__((ext_vector_type(4)));

typedef unsigned int u32x4 __attribute__((ext_vector_type(4)));
typedef int          i32x4 __attribute__((ext_vector_type(4)));
typedef int          i32x8 __attribute__((ext_vector_type(8)));

#if __has_builtin(__builtin_amdgcn_tensor_load_to_lds) && \
    __has_builtin(__builtin_amdgcn_s_wait_tensorcnt)
#define USE_TDM 1
#else
#define USE_TDM 0
#endif

// ---------------- workspace layout (bytes) ----------------
#define OFF_FMASK  0UL          // 16*512*512       = 4 MiB  (bytes 0/1)
#define OFF_BMASK  4194304UL    // 4 MiB
#define OFF_HF     8388608UL    // h state  f32  (16*512*256) = 8 MiB
#define OFF_CF     16777216UL   // c state  f32  8 MiB
#define OFF_XMBF   25165824UL   // xm bf16  (8192*256)        = 4 MiB
#define OFF_XPART  29360128UL   // x-part f32 [i][b][1024]    = 32 MiB
#define OFF_WXF    62914560UL   // Wx fragments bf16 (64*8*32*16) = 512 KiB
#define OFF_WHF    63438848UL   // Wh fragments bf16 512 KiB
#define OFF_BIAS   63963136UL   // combined bias f32 [1024]
#define OFF_FCNT   63967232UL   // int [8192]
#define OFF_BCNT   64000000UL   // int [8192]
#define OFF_FLIST  64032768UL   // int [8192][CAP]
#define OFF_BLIST  65605632UL   // int [8192][CAP]
#define OFF_CTR    67178496UL   // unsigned barrier counter

__device__ __forceinline__ float sigf(float x) { return 1.0f / (1.0f + __expf(-x)); }

// ---------------------------------------------------------------------------
// 0) zero masks + h/c state + barrier counter (required every launch)
// ---------------------------------------------------------------------------
__global__ void k_zero(f4* p, long n16, unsigned* ctr) {
    long i = blockIdx.x * (long)blockDim.x + threadIdx.x;
    long stride = (long)gridDim.x * blockDim.x;
    f4 z = {0.f, 0.f, 0.f, 0.f};
    for (; i < n16; i += stride) p[i] = z;
    if (blockIdx.x == 0 && threadIdx.x == 0) *ctr = 0u;
}

// ---------------------------------------------------------------------------
// 1) xm = mean_t emb[x], stored bf16 row-major (row = b*512+n)
// ---------------------------------------------------------------------------
__global__ void k_embed(const int* __restrict__ x, const float* __restrict__ emb,
                        bf16* __restrict__ xm) {
    int row = blockIdx.x;        // 0..8191
    int h   = threadIdx.x;       // 0..255
    const int* xt = x + row * Tt;
    float s = 0.f;
    #pragma unroll
    for (int t = 0; t < Tt; ++t) s += emb[xt[t] * Hh + h];
    xm[row * Hh + h] = (bf16)(s * (1.0f / 16.0f));
}

// ---------------------------------------------------------------------------
// 2) scatter edges -> byte masks (idempotent writes of 1 => set() semantics)
// ---------------------------------------------------------------------------
__global__ void k_scatter(const int* __restrict__ fe, const int* __restrict__ be,
                          unsigned char* __restrict__ fm, unsigned char* __restrict__ bm) {
    int id = blockIdx.x * 256 + threadIdx.x;   // 0..16383 = b*1024+e
    int b = id >> 10, e = id & 1023;
    int base = (b * Ee + e) * 2;
    int s = fe[base] - 1, d = fe[base + 1] - 1;
    fm[(size_t)b * Nn * Nn + d * Nn + s] = 1;
    s = be[base] - 1; d = be[base + 1] - 1;
    bm[(size_t)b * Nn * Nn + d * Nn + s] = 1;
}

// ---------------------------------------------------------------------------
// 3) build CSR-ish predecessor lists (dedup via mask); mat[:,0,0] cleared here
// ---------------------------------------------------------------------------
__global__ void k_lists(const unsigned char* __restrict__ fm, const unsigned char* __restrict__ bm,
                        int* __restrict__ fcnt, int* __restrict__ flist,
                        int* __restrict__ bcnt, int* __restrict__ blist) {
    int id = blockIdx.x * 256 + threadIdx.x;   // 0..8191 = b*512+i
    int b = id >> 9, i = id & 511;
    const unsigned char* fr = fm + (size_t)b * Nn * Nn + (size_t)i * Nn;
    const unsigned char* br = bm + (size_t)b * Nn * Nn + (size_t)i * Nn;
    int cf = 0, cb = 0;
    for (int n = 0; n < Nn; ++n) {
        bool z00 = (i == 0) && (n == 0);       // mat.at[:,0,0].set(0)
        if (fr[n] && !z00 && cf < CAP) flist[(size_t)id * CAP + cf++] = n;
        if (br[n] && !z00 && cb < CAP) blist[(size_t)id * CAP + cb++] = n;
    }
    fcnt[id] = cf;  bcnt[id] = cb;
}

// ---------------------------------------------------------------------------
// 4) pack weights into exact WMMA B-fragment lane layout (bf16), fold biases.
//    Combined column space: [0,768)=i/o/u gate rows, [768,1024)=f gate.
//    B[k][n] = W[col=n][k]; lane holds n=lane&15, K per ISA 16-bit layout:
//    k = kt*32 + ((lane&16)?8:0) + (e<8 ? e : e+8)
// ---------------------------------------------------------------------------
__global__ void k_prepw(const float* __restrict__ W_ax, const float* __restrict__ b_ax,
                        const float* __restrict__ W_ah, const float* __restrict__ b_ah,
                        const float* __restrict__ W_fx, const float* __restrict__ b_fx,
                        const float* __restrict__ W_fh, const float* __restrict__ b_fh,
                        bf16* __restrict__ wxf, bf16* __restrict__ whf,
                        float* __restrict__ bias) {
    int tid  = blockIdx.x * 256 + threadIdx.x;   // 0..16383 = tn*256 + kt*32 + lane
    int tn   = tid >> 8;
    int kt   = (tid >> 5) & 7;
    int lane = tid & 31;
    int cb   = (lane & 16) ? 8 : 0;
    int col  = tn * 16 + (lane & 15);
    #pragma unroll
    for (int e = 0; e < 16; ++e) {
        int k = kt * 32 + cb + (e < 8 ? e : e + 8);
        float wh = (col < 768) ? W_ah[col * Hh + k] : W_fh[(col - 768) * Hh + k];
        float wx = (col < 768) ? W_ax[col * Hh + k] : W_fx[(col - 768) * Hh + k];
        size_t idx = ((size_t)(tn * 8 + kt) * 32 + lane) * 16 + e;
        whf[idx] = (bf16)wh;
        wxf[idx] = (bf16)wx;
    }
    if (kt == 0 && lane < 16) {
        bias[col] = (col < 768) ? (b_ax[col] + b_ah[col])
                                : (b_fx[col - 768] + b_fh[col - 768]);
    }
}

// ---------------------------------------------------------------------------
// 5) big x-part GEMM with WMMA:  xpart[i][b][col] = xm @ [W_ax;W_fx]^T + bias
//    grid(512, 16), block 128 (4 waves); wave w -> ntile = blockIdx.y*4 + w
// ---------------------------------------------------------------------------
__global__ __launch_bounds__(128) void k_xgemm(const bf16* __restrict__ xm,
                                               const bf16* __restrict__ wxf,
                                               const float* __restrict__ bias,
                                               float* __restrict__ xpart) {
    int mtile = blockIdx.x;                        // 0..511
    int wave  = threadIdx.x >> 5, lane = threadIdx.x & 31;
    int tn    = blockIdx.y * 4 + wave;             // 0..63
    int cb    = (lane & 16) ? 8 : 0;
    int arow  = mtile * 16 + (lane & 15);          // global row = b*512+pos

    v8f acc;
    float bv = bias[tn * 16 + (lane & 15)];
    #pragma unroll
    for (int r = 0; r < 8; ++r) acc[r] = bv;

    #pragma unroll
    for (int kt = 0; kt < 8; ++kt) {
        union { f4 q[2]; v16bf v; } ua;
        const f4* p = (const f4*)(xm + (size_t)arow * Hh + kt * 32 + cb);
        ua.q[0] = p[0];                // K [kb, kb+8)
        ua.q[1] = p[2];                // K [kb+16, kb+24)
        v16bf bfr = *(const v16bf*)(wxf + ((size_t)(tn * 8 + kt) * 32 + lane) * 16);
        acc = __builtin_amdgcn_wmma_f32_16x16x32_bf16(false, ua.v, false, bfr,
                                                      (short)0, acc, false, false);
    }

    int ncol = tn * 16 + (lane & 15);
    #pragma unroll
    for (int r = 0; r < 8; ++r) {
        int mrow = mtile * 16 + r + ((lane >> 4) & 1) * 8;   // b*512+pos
        int b = mrow >> 9, pos = mrow & 511;
        xpart[((size_t)pos * 16 + b) * 1024 + ncol] = acc[r];
    }
}

// ---------------------------------------------------------------------------
// 6) recurrent scan: 16 cooperating workgroups (block w owns h dims [16w,16w+16),
//    its 4 waves compute gates i/o/u/f as WMMA 16x16 tiles with register-resident
//    B fragments). Per-step the x-part tile arrives via TDM (tensor_load_to_lds,
//    overlapped with the sparse h-gather, completed by s_wait_tensorcnt).
//    Device-wide barrier per step (cluster-barrier + atomic + acquire fence).
// ---------------------------------------------------------------------------
__global__ __launch_bounds__(128) void k_recur(const float* __restrict__ xpart,
                                               const bf16* __restrict__ whf,
                                               const int* __restrict__ fcnt,
                                               const int* __restrict__ flist,
                                               float* __restrict__ h_f,
                                               float* __restrict__ c_f,
                                               unsigned* __restrict__ ctr) {
    __shared__ bf16  lds_h[16 * 256];   // h_last (A matrix), bf16
    __shared__ float lds_c[16 * 16];    // c_last, own slice
    __shared__ float lds_g[4 * 16 * 16];// gate pre-activations i/o/u/f
#if USE_TDM
    __shared__ float lds_x[4 * 16 * 16];// per-wave x-part tile landed by TDM
#endif

    const int w    = blockIdx.x;          // 0..15
    const int t    = threadIdx.x;         // 0..127
    const int lane = t & 31, wave = t >> 5;
    const int tn   = wave * 16 + w;       // weight tile: gate=wave, slice=w
    const int cb   = (lane & 16) ? 8 : 0;

    // recurrent-weight fragments stay in registers for all 512 steps
    v16bf bfrag[8];
    #pragma unroll
    for (int kt = 0; kt < 8; ++kt)
        bfrag[kt] = *(const v16bf*)(whf + ((size_t)(tn * 8 + kt) * 32 + lane) * 16);

#if USE_TDM
    // static D# group1: data_size=4B, tensor_dim0=1024, tensor_dim1=8192,
    // tile 16x16, dim0_stride=1024 elements (see CDNA5 ISA bitfields)
    i32x8 g1;
    g1[0] = 0x00020000;   // workgroup_mask=0 | data_size=2 (4B) <<16
    g1[1] = 0x04000000;   // tensor_dim0[15:0]=1024 in bits[31:16]
    g1[2] = 0x20000000;   // tensor_dim1[15:0]=8192 in bits[31:16] | dim0[31:16]=0
    g1[3] = 0x00100000;   // tile_dim0=16 in bits[31:16] | dim1[31:16]=0
    g1[4] = 16;           // tile_dim1=16, tile_dim2=0
    g1[5] = 1024;         // tensor_dim0_stride[31:0]
    g1[6] = 0;            // stride[47:32]=0, dim1_stride low bits 0
    g1[7] = 0;
    const i32x4 gz = {0, 0, 0, 0};
    typedef __attribute__((address_space(3))) float as3f;
    unsigned lds_x_off = (unsigned)(size_t)(as3f*)(&lds_x[wave * 256]);
#endif

    for (int i = 0; i < Nn; ++i) {
#if USE_TDM
        // ---- kick the TDM: 16x16 f32 tile (rows i*16.., cols tn*16..) -> LDS ----
        {
            unsigned long long ga =
                (unsigned long long)(const void*)(xpart + (size_t)i * 16 * 1024 + tn * 16);
            u32x4 g0;
            g0[0] = 1u;                                   // count=1 (valid user D#)
            g0[1] = lds_x_off;                            // lds_addr (bytes)
            g0[2] = (unsigned)(ga & 0xFFFFFFFFu);         // global_addr[31:0]
            g0[3] = (unsigned)((ga >> 32) & 0x01FFFFFFu)  // global_addr[56:32]
                    | 0x80000000u;                        // type=2 ("image")
#if defined(__clang_major__) && (__clang_major__ >= 23)
            const i32x8 gz8 = {0, 0, 0, 0, 0, 0, 0, 0};
            __builtin_amdgcn_tensor_load_to_lds(g0, g1, gz, gz, gz8, 0);
#else
            __builtin_amdgcn_tensor_load_to_lds(g0, g1, gz, gz, 0);
#endif
        }
#else
        // prefetch next step's x-part block into L2 (global_prefetch_b8)
        int inext = (i + 1 < Nn) ? i + 1 : i;
        __builtin_prefetch(xpart + (size_t)inext * 16 * 1024 + t * 128, 0, 1);
#endif

        // ---- sparse gather: h_last (full 16x256) and c_last (own 16x16) ----
        {
            int c0 = t * 2;                                 // 2 columns per thread
            for (int b = 0; b < Bb; ++b) {
                int ri  = b * Nn + i;
                int cnt = fcnt[ri];
                float s0 = 0.f, s1 = 0.f;
                for (int p = 0; p < cnt; ++p) {
                    int n = flist[(size_t)ri * CAP + p];
                    const float* hr = h_f + ((size_t)b * Nn + n) * Hh;
                    s0 += hr[c0]; s1 += hr[c0 + 1];
                }
                lds_h[b * Hh + c0]     = (bf16)s0;
                lds_h[b * Hh + c0 + 1] = (bf16)s1;
            }
            for (int e = t; e < 256; e += 128) {            // c_last own slice
                int b = e >> 4, jj = e & 15;
                int ri  = b * Nn + i;
                int cnt = fcnt[ri];
                float s = 0.f;
                for (int p = 0; p < cnt; ++p) {
                    int n = flist[(size_t)ri * CAP + p];
                    s += c_f[((size_t)b * Nn + n) * Hh + w * 16 + jj];
                }
                lds_c[e] = s;
            }
        }
        __syncthreads();

        // ---- WMMA: acc = x_part tile + h_last @ W_gate^T ----
        v8f acc;
#if USE_TDM
        __builtin_amdgcn_s_wait_tensorcnt(0);   // tile landed in LDS
        #pragma unroll
        for (int r = 0; r < 8; ++r) {
            int m = r + ((lane >> 4) & 1) * 8;              // batch index
            acc[r] = lds_x[wave * 256 + m * 16 + (lane & 15)];
        }
#else
        int ncol0 = tn * 16 + (lane & 15);
        #pragma unroll
        for (int r = 0; r < 8; ++r) {
            int m = r + ((lane >> 4) & 1) * 8;              // batch index
            acc[r] = xpart[((size_t)i * 16 + m) * 1024 + ncol0];
        }
#endif
        #pragma unroll
        for (int kt = 0; kt < 8; ++kt) {
            int row = lane & 15;
            const f4* p = (const f4*)&lds_h[row * Hh + kt * 32 + cb];
            union { f4 q[2]; v16bf v; } ua;
            ua.q[0] = p[0]; ua.q[1] = p[2];
            acc = __builtin_amdgcn_wmma_f32_16x16x32_bf16(false, ua.v, false, bfrag[kt],
                                                          (short)0, acc, false, false);
        }
        #pragma unroll
        for (int r = 0; r < 8; ++r) {
            int m = r + ((lane >> 4) & 1) * 8;
            lds_g[wave * 256 + m * 16 + (lane & 15)] = acc[r];
        }
        __syncthreads();

        // ---- nonlinearities + state update for own 16 dims ----
        for (int e = t; e < 256; e += 128) {
            int b = e >> 4, jj = e & 15;
            float ig = lds_g[0 * 256 + b * 16 + jj];
            float og = lds_g[1 * 256 + b * 16 + jj];
            float ug = lds_g[2 * 256 + b * 16 + jj];
            float fg = lds_g[3 * 256 + b * 16 + jj];
            float cl = lds_c[e];
            float c  = sigf(ig) * tanhf(ug) + sigf(fg) * cl;
            float h  = sigf(og) * tanhf(c);
            int col  = w * 16 + jj;
            h_f[((size_t)b * Nn + i) * Hh + col] = h;
            c_f[((size_t)b * Nn + i) * Hh + col] = c;
        }

        // ---- device-wide step barrier ----
        __threadfence();                       // release h/c stores to device scope
        __syncthreads();
        if (t == 0) {
            __builtin_amdgcn_s_cluster_barrier();    // CDNA5 cluster barrier (NOP if uncluster)
            atomicAdd(ctr, 1u);
            unsigned target = (unsigned)(16 * (i + 1));
            while (__hip_atomic_load(ctr, __ATOMIC_ACQUIRE, __HIP_MEMORY_SCOPE_AGENT) < target)
                __builtin_amdgcn_s_sleep(2);
        }
        __syncthreads();
        __builtin_amdgcn_fence(__ATOMIC_ACQUIRE, "agent");  // invalidate for fresh h/c reads
    }
}

// ---------------------------------------------------------------------------
// 7) back pass: each (b,h) chain is independent -> 4096 sequential lanes, no sync
// ---------------------------------------------------------------------------
__global__ void k_back(const int* __restrict__ bcnt, const int* __restrict__ blist,
                       float* __restrict__ h_f) {
    int id = blockIdx.x * blockDim.x + threadIdx.x;   // 0..4095
    int b = id >> 8, h = id & 255;
    float* hb = h_f + (size_t)b * Nn * Hh;
    for (int j = 0; j < Nn; ++j) {
        int ri  = b * Nn + j;
        int cnt = bcnt[ri];
        float s = 0.f;
        for (int p = 0; p < cnt; ++p)
            s += hb[(size_t)blist[(size_t)ri * CAP + p] * Hh + h];
        hb[(size_t)j * Hh + h] += s;
    }
}

// ---------------------------------------------------------------------------
// 8) output: sigmoid(h_f @ W_out^T + b_out), one block (256 thr) per row
// ---------------------------------------------------------------------------
__global__ void k_out(const float* __restrict__ h_f, const float* __restrict__ W_out,
                      const float* __restrict__ b_out, float* __restrict__ out) {
    __shared__ float red[256];
    int row = blockIdx.x, t = threadIdx.x;
    red[t] = h_f[(size_t)row * Hh + t] * W_out[t];
    __syncthreads();
    for (int s = 128; s > 0; s >>= 1) {
        if (t < s) red[t] += red[t + s];
        __syncthreads();
    }
    if (t == 0) out[row] = sigf(red[0] + b_out[0]);
}

// ---------------------------------------------------------------------------
extern "C" void kernel_launch(void* const* d_in, const int* in_sizes, int n_in,
                              void* d_out, int out_size, void* d_ws, size_t ws_size,
                              hipStream_t stream) {
    const int*   x      = (const int*)  d_in[0];
    const int*   fedges = (const int*)  d_in[1];
    const int*   bedges = (const int*)  d_in[2];
    const float* emb    = (const float*)d_in[3];
    const float* W_ax   = (const float*)d_in[4];
    const float* b_ax   = (const float*)d_in[5];
    const float* W_ah   = (const float*)d_in[6];
    const float* b_ah   = (const float*)d_in[7];
    const float* W_fx   = (const float*)d_in[8];
    const float* b_fx   = (const float*)d_in[9];
    const float* W_fh   = (const float*)d_in[10];
    const float* b_fh   = (const float*)d_in[11];
    const float* W_out  = (const float*)d_in[12];
    const float* b_out  = (const float*)d_in[13];
    float* out = (float*)d_out;

    char* ws = (char*)d_ws;
    unsigned char* fmask = (unsigned char*)(ws + OFF_FMASK);
    unsigned char* bmask = (unsigned char*)(ws + OFF_BMASK);
    float* h_f   = (float*)(ws + OFF_HF);
    float* c_f   = (float*)(ws + OFF_CF);
    bf16*  xm    = (bf16*) (ws + OFF_XMBF);
    float* xpart = (float*)(ws + OFF_XPART);
    bf16*  wxf   = (bf16*) (ws + OFF_WXF);
    bf16*  whf   = (bf16*) (ws + OFF_WHF);
    float* bias  = (float*)(ws + OFF_BIAS);
    int*   fcnt  = (int*)  (ws + OFF_FCNT);
    int*   bcnt  = (int*)  (ws + OFF_BCNT);
    int*   flist = (int*)  (ws + OFF_FLIST);
    int*   blist = (int*)  (ws + OFF_BLIST);
    unsigned* ctr = (unsigned*)(ws + OFF_CTR);

    // 0) zero masks + h/c state + barrier counter (25165824 bytes = 1572864 f4)
    k_zero<<<1024, 256, 0, stream>>>((f4*)ws, 1572864L, ctr);
    // 1) embedding mean -> bf16
    k_embed<<<Bb * Nn, Hh, 0, stream>>>(x, emb, xm);
    // 2) edge scatter into byte masks
    k_scatter<<<64, 256, 0, stream>>>(fedges, bedges, fmask, bmask);
    // 3) dedup'd predecessor lists
    k_lists<<<32, 256, 0, stream>>>(fmask, bmask, fcnt, flist, bcnt, blist);
    // 4) weight fragment packing + bias folding
    k_prepw<<<64, 256, 0, stream>>>(W_ax, b_ax, W_ah, b_ah, W_fx, b_fx, W_fh, b_fh,
                                    wxf, whf, bias);
    // 5) x-part GEMM (WMMA)
    k_xgemm<<<dim3(512, 16), 128, 0, stream>>>(xm, wxf, bias, xpart);
    // 6) recurrent scan (WMMA + TDM tile feed + device-wide step barrier)
    k_recur<<<16, 128, 0, stream>>>(xpart, whf, fcnt, flist, h_f, c_f, ctr);
    // 7) back pass (independent (b,h) chains)
    k_back<<<16, 256, 0, stream>>>(bcnt, blist, h_f);
    // 8) output projection + sigmoid
    k_out<<<Bb * Nn, Hh, 0, stream>>>(h_f, W_out, b_out, out);
}
